// SelfAttention_55911884259785
// MI455X (gfx1250) — matrix operand order repaired and checked by reading, not compiled, not run on hardware
//
#include <hip/hip_runtime.h>
#include <hip/hip_bf16.h>

#define DIM    1024
#define HEADS  16
#define HD     64          // head dim
#define NB     2           // batch
#define T      1024        // seq len
#define ROWS   (NB * T)    // 2048
#define CHUNK  64
#define NCHUNK (T / CHUNK) // 16
#define EPSF   1e-6f

typedef __attribute__((ext_vector_type(16))) __bf16 v16bf;
typedef __attribute__((ext_vector_type(8)))  __bf16 v8bf;
typedef __attribute__((ext_vector_type(8)))  float  v8f;

// ---------------------------------------------------------------------------
// fp32 -> bf16 elementwise convert
// ---------------------------------------------------------------------------
__global__ __launch_bounds__(256) void cvt_bf16(const float* __restrict__ s,
                                                __bf16* __restrict__ d, int n) {
  int i = blockIdx.x * blockDim.x + threadIdx.x;
  if (i < n) d[i] = (__bf16)s[i];
}

// fp32 W[k][n] -> bf16 Wt[n][k]  (transpose so WMMA B-fragments are contiguous)
__global__ __launch_bounds__(256) void cvt_bf16_t(const float* __restrict__ W,
                                                  __bf16* __restrict__ Wt) {
  int i = blockIdx.x * blockDim.x + threadIdx.x;   // 0 .. DIM*DIM-1
  int n = i & (DIM - 1);
  int k = i >> 10;
  Wt[(size_t)n * DIM + k] = (__bf16)W[(size_t)k * DIM + n];
}

// ---------------------------------------------------------------------------
// C[2048][1024] = A[2048][1024](bf16) @ Bt[1024][1024](bf16)^T (+ bias),
// f32 accumulate, WMMA. Block = 8 waves -> 128x64 tile. The shared B slab
// (64x32 bf16) is staged into a double-buffered, pad-striped LDS tile with
// global_load_async_to_lds_b128 (ASYNCcnt), overlapping the fill of slab i+1
// with the WMMAs on slab i. A fragments are software-pipelined in registers.
// Per k-step: batch all 8 ds_load_b128 first, then 4 back-to-back WMMAs so a
// single s_wait_dscnt covers the whole batch.
// ---------------------------------------------------------------------------
#define GM   ROWS           // 2048
#define GN   DIM            // 1024
#define GK   DIM            // 1024
#define N64  (GN / 64)      // 16
#define NKS  (GK / 32)      // 32
#define BPAD 40             // LDS B row stride (bf16): 80 B -> conflict-free
__global__ __launch_bounds__(256) void gemm_bf16(const __bf16* __restrict__ A,
                                                 const __bf16* __restrict__ Bt,
                                                 const float* __restrict__ bias,
                                                 float* __restrict__ C) {
  __shared__ __bf16 Bs[2][64][BPAD];

  const int lane = threadIdx.x & 31;
  const int wave = threadIdx.x >> 5;
  const int mt   = blockIdx.x / N64;       // 128-row block
  const int nt   = blockIdx.x % N64;       // 64-col block
  const int n0   = nt * 64;

  // cooperative B-slab fill: thread -> (row r of 64, 16-byte chunk ch of 4)
  const int r  = threadIdx.x >> 2;
  const int ch = threadIdx.x & 3;
  const __bf16* __restrict__ fill_gp = Bt + (size_t)(n0 + r) * GK + ch * 8;
  const unsigned fill_lds =
      (unsigned)(uintptr_t)&Bs[0][r][ch * 8];          // buf 0 LDS byte offset
  const unsigned buf_stride = (unsigned)sizeof(__bf16) * 64 * BPAD;

  auto fill = [&](int buf, int k0) {
    unsigned long long ga = (unsigned long long)(uintptr_t)(fill_gp + k0);
    unsigned la = fill_lds + (unsigned)buf * buf_stride;
    asm volatile("global_load_async_to_lds_b128 %0, %1, off"
                 :: "v"(la), "v"(ga) : "memory");
  };

  // A fragment addressing (register double buffer)
  const int arow  = mt * 128 + wave * 16 + (lane & 15);
  const int akoff = (lane >> 4) * 8;       // K halves {0..7,16..23}/{8..15,24..31}
  const __bf16* __restrict__ ap = A + (size_t)arow * GK + akoff;

  union AF { v16bf v; v8bf h[2]; };
  AF a_cur, a_nxt;
  a_cur.h[0] = *(const v8bf*)(ap + 0);
  a_cur.h[1] = *(const v8bf*)(ap + 16);
  fill(0, 0);

  v8f acc[4] = {};
  for (int ks = 0; ks < NKS; ++ks) {
    const int k0 = ks << 5;
    if (ks + 1 < NKS) {
      fill((ks + 1) & 1, k0 + 32);                       // async fill next slab
      a_nxt.h[0] = *(const v8bf*)(ap + k0 + 32);
      a_nxt.h[1] = *(const v8bf*)(ap + k0 + 48);
      __builtin_prefetch(ap + k0 + 256, 0, 0);           // global_prefetch_b8
      asm volatile("s_wait_asynccnt 0x1" ::: "memory");  // my prior fill done
    } else {
      asm volatile("s_wait_asynccnt 0x0" ::: "memory");
    }
    __syncthreads();                    // slab `cur` fully resident for all waves

    const int cur = ks & 1;
    union BF { v16bf v; v8bf h[2]; } bfrag[4];
#pragma unroll
    for (int j = 0; j < 4; ++j) {       // batch all 8 ds_load_b128 first
      const __bf16* bp = &Bs[cur][16 * j + (lane & 15)][(lane >> 4) * 8];
      bfrag[j].h[0] = *(const v8bf*)bp;
      bfrag[j].h[1] = *(const v8bf*)(bp + 16);
    }
#pragma unroll
    for (int j = 0; j < 4; ++j)         // then 4 back-to-back WMMAs
      acc[j] = __builtin_amdgcn_wmma_f32_16x16x32_bf16(
          false, a_cur.v, false, bfrag[j].v, (short)0, acc[j], false, false);

    if (ks + 1 < NKS) a_cur = a_nxt;
    __syncthreads();                    // slab `cur` free for refill at ks+2
  }

  const int rbase = mt * 128 + wave * 16 + (lane >> 4) * 8;
  const int cbase = n0 + (lane & 15);
#pragma unroll
  for (int j = 0; j < 4; ++j) {
    const int col = cbase + 16 * j;
    const float bv = bias ? bias[col] : 0.0f;
#pragma unroll
    for (int i = 0; i < 8; ++i)
      C[(size_t)(rbase + i) * GN + col] = acc[j][i] + bv;
  }
}

// ---------------------------------------------------------------------------
// Per (row,head): softmax over d=64 on Q (in place); psi(k)=elu(k)+1 on K.
// (row*DIM + h*HD) == tid*64, so each thread owns 64 contiguous floats.
// ---------------------------------------------------------------------------
__global__ __launch_bounds__(256) void qk_transform(float* __restrict__ Q,
                                                    float* __restrict__ Kp) {
  const int tid = blockIdx.x * blockDim.x + threadIdx.x;   // 0..ROWS*HEADS-1
  float* __restrict__ q = Q  + (size_t)tid * HD;
  float* __restrict__ k = Kp + (size_t)tid * HD;
  float m = q[0];
  for (int d = 1; d < HD; ++d) m = fmaxf(m, q[d]);
  float s = 0.f;
  for (int d = 0; d < HD; ++d) s += __expf(q[d] - m);
  const float inv = 1.0f / s;
  for (int d = 0; d < HD; ++d) q[d] = __expf(q[d] - m) * inv;
  for (int d = 0; d < HD; ++d) {
    float x = k[d];
    k[d] = (x > 0.f) ? (x + 1.0f) : __expf(x);             // elu(x)+1
  }
}

// ---------------------------------------------------------------------------
// Pass 1: per-chunk summaries. block = (b,h,c): KV_c = K_chunk^T V_chunk,
// Ksum_c[d] = sum_t k[t][d].  Thread (g=tid>>6, e=tid&63) owns S[ds..ds+15][e].
// ---------------------------------------------------------------------------
__global__ __launch_bounds__(256) void chunk_sums(const float* __restrict__ Kp,
                                                  const float* __restrict__ V,
                                                  float* __restrict__ KV,
                                                  float* __restrict__ Ksum) {
  __shared__ float Kl[CHUNK][HD];
  __shared__ float Vl[CHUNK][HD];
  const int c = blockIdx.x & 15, h = (blockIdx.x >> 4) & 15, b = blockIdx.x >> 8;
  const size_t gbase = ((size_t)(b * T + c * CHUNK)) * DIM + h * HD;

  for (int it = 0; it < 16; ++it) {
    int idx = it * 256 + threadIdx.x;
    int tl = idx >> 6, d = idx & 63;
    Kl[tl][d] = Kp[gbase + (size_t)tl * DIM + d];
    Vl[tl][d] = V [gbase + (size_t)tl * DIM + d];
  }
  __syncthreads();

  const int e = threadIdx.x & 63, ds = (threadIdx.x >> 6) * 16;
  float acc[16];
#pragma unroll
  for (int i = 0; i < 16; ++i) acc[i] = 0.f;
  for (int t = 0; t < CHUNK; ++t) {
    const float vv = Vl[t][e];
#pragma unroll
    for (int i = 0; i < 16; ++i) acc[i] += Kl[t][ds + i] * vv;
  }
  const size_t kvb = (size_t)blockIdx.x * (HD * HD);
#pragma unroll
  for (int i = 0; i < 16; ++i) KV[kvb + (size_t)(ds + i) * HD + e] = acc[i];

  if (threadIdx.x < HD) {
    float s = 0.f;
    for (int t = 0; t < CHUNK; ++t) s += Kl[t][threadIdx.x];
    Ksum[(size_t)blockIdx.x * HD + threadIdx.x] = s;
  }
}

// ---------------------------------------------------------------------------
// Pass 2: exclusive scan over the 16 chunks per (b,h), in place. block=(b,h).
// ---------------------------------------------------------------------------
__global__ __launch_bounds__(256) void chunk_scan(float* __restrict__ KV,
                                                  float* __restrict__ Ksum) {
  const int bh = blockIdx.x;                      // 0..31 = b*16+h
  const int e = threadIdx.x & 63, ds = (threadIdx.x >> 6) * 16;
  float run[16] = {};
  for (int c = 0; c < NCHUNK; ++c) {
    const size_t base = ((size_t)bh * NCHUNK + c) * (HD * HD);
#pragma unroll
    for (int i = 0; i < 16; ++i) {
      const size_t o = base + (size_t)(ds + i) * HD + e;
      const float prev = run[i];
      run[i] += KV[o];
      KV[o] = prev;                               // exclusive prefix
    }
  }
  if (threadIdx.x < HD) {
    float r = 0.f;
    for (int c = 0; c < NCHUNK; ++c) {
      const size_t o = ((size_t)bh * NCHUNK + c) * HD + threadIdx.x;
      const float prev = r;
      r += Ksum[o];
      Ksum[o] = prev;
    }
  }
}

// ---------------------------------------------------------------------------
// Pass 3: intra-chunk recurrence seeded with (Sprev, zprev). block=(b,h,c).
// out_t[e] = sum_d q_t[d]/(z_t[d]+eps) * S_t[d][e], written as bf16 (b,t,e).
// ---------------------------------------------------------------------------
__global__ __launch_bounds__(256) void intra_chunk(const float* __restrict__ Qs,
                                                   const float* __restrict__ Kp,
                                                   const float* __restrict__ V,
                                                   const float* __restrict__ KV,
                                                   const float* __restrict__ Ksum,
                                                   __bf16* __restrict__ attn) {
  __shared__ float Ql[CHUNK][HD];
  __shared__ float Kl[CHUNK][HD];
  __shared__ float Vl[CHUNK][HD];
  __shared__ float partial[4][HD];
  const int c = blockIdx.x & 15, h = (blockIdx.x >> 4) & 15, b = blockIdx.x >> 8;
  const size_t gbase = ((size_t)(b * T + c * CHUNK)) * DIM + h * HD;

  for (int it = 0; it < 16; ++it) {
    int idx = it * 256 + threadIdx.x;
    int tl = idx >> 6, d = idx & 63;
    Ql[tl][d] = Qs[gbase + (size_t)tl * DIM + d];
    Kl[tl][d] = Kp[gbase + (size_t)tl * DIM + d];
    Vl[tl][d] = V [gbase + (size_t)tl * DIM + d];
  }

  const int e = threadIdx.x & 63, g = threadIdx.x >> 6, ds = g * 16;
  float S[16], zr[16];
  const size_t kvb = (size_t)blockIdx.x * (HD * HD);
#pragma unroll
  for (int i = 0; i < 16; ++i) S[i] = KV[kvb + (size_t)(ds + i) * HD + e];
#pragma unroll
  for (int i = 0; i < 16; ++i) zr[i] = Ksum[(size_t)blockIdx.x * HD + ds + i];
  __syncthreads();

  for (int t = 0; t < CHUNK; ++t) {
    const float vv = Vl[t][e];
    float p = 0.f;
#pragma unroll
    for (int i = 0; i < 16; ++i) {
      const float kv = Kl[t][ds + i];
      zr[i] += kv;                                // inclusive z_t
      S[i]  += kv * vv;                           // inclusive S_t
      p += Ql[t][ds + i] * S[i] / (zr[i] + EPSF);
    }
    partial[g][e] = p;
    __syncthreads();
    if (g == 0) {
      const float o = partial[0][e] + partial[1][e] + partial[2][e] + partial[3][e];
      attn[gbase + (size_t)t * DIM + e] = (__bf16)o;
    }
    __syncthreads();
  }
}

// ---------------------------------------------------------------------------
extern "C" void kernel_launch(void* const* d_in, const int* in_sizes, int n_in,
                              void* d_out, int out_size, void* d_ws, size_t ws_size,
                              hipStream_t stream) {
  const float* x  = (const float*)d_in[0];
  const float* Wq = (const float*)d_in[1];
  const float* Wk = (const float*)d_in[2];
  const float* Wv = (const float*)d_in[3];
  const float* Wo = (const float*)d_in[4];
  const float* bo = (const float*)d_in[5];
  float* out = (float*)d_out;

  // workspace carve (256-B aligned)
  char* w = (char*)d_ws;
  auto carve = [&](size_t bytes) -> void* {
    void* p = (void*)w;
    w += (bytes + 255) & ~(size_t)255;
    return p;
  };
  __bf16* xb   = (__bf16*)carve((size_t)ROWS * DIM * sizeof(__bf16));
  __bf16* Wqt  = (__bf16*)carve((size_t)DIM * DIM * sizeof(__bf16));
  __bf16* Wkt  = (__bf16*)carve((size_t)DIM * DIM * sizeof(__bf16));
  __bf16* Wvt  = (__bf16*)carve((size_t)DIM * DIM * sizeof(__bf16));
  __bf16* Wot  = (__bf16*)carve((size_t)DIM * DIM * sizeof(__bf16));
  float*  Qr   = (float*)carve((size_t)ROWS * DIM * sizeof(float));
  float*  Kr   = (float*)carve((size_t)ROWS * DIM * sizeof(float));
  float*  Vr   = (float*)carve((size_t)ROWS * DIM * sizeof(float));
  float*  KV   = (float*)carve((size_t)NB * HEADS * NCHUNK * HD * HD * sizeof(float));
  float*  Ksum = (float*)carve((size_t)NB * HEADS * NCHUNK * HD * sizeof(float));
  __bf16* attn = (__bf16*)carve((size_t)ROWS * DIM * sizeof(__bf16));

  // 1) convert inputs to bf16 (weights transposed for contiguous B fragments)
  cvt_bf16<<<(ROWS * DIM) / 256, 256, 0, stream>>>(x, xb, ROWS * DIM);
  cvt_bf16_t<<<(DIM * DIM) / 256, 256, 0, stream>>>(Wq, Wqt);
  cvt_bf16_t<<<(DIM * DIM) / 256, 256, 0, stream>>>(Wk, Wkt);
  cvt_bf16_t<<<(DIM * DIM) / 256, 256, 0, stream>>>(Wv, Wvt);
  cvt_bf16_t<<<(DIM * DIM) / 256, 256, 0, stream>>>(Wo, Wot);

  // 2) Q/K/V projections via WMMA   (grid = M/128 * N/64 blocks of 8 waves)
  const int gemm_blocks = (GM / 128) * (GN / 64);        // 256
  gemm_bf16<<<gemm_blocks, 256, 0, stream>>>(xb, Wqt, nullptr, Qr);
  gemm_bf16<<<gemm_blocks, 256, 0, stream>>>(xb, Wkt, nullptr, Kr);
  gemm_bf16<<<gemm_blocks, 256, 0, stream>>>(xb, Wvt, nullptr, Vr);

  // 3) softmax(q) over head dim; psi(k)=elu+1 (in place)
  qk_transform<<<(ROWS * HEADS) / 256, 256, 0, stream>>>(Qr, Kr);

  // 4) chunked linear attention: summaries -> exclusive scan -> intra-chunk
  chunk_sums<<<NB * HEADS * NCHUNK, 256, 0, stream>>>(Kr, Vr, KV, Ksum);
  chunk_scan<<<NB * HEADS, 256, 0, stream>>>(KV, Ksum);
  intra_chunk<<<NB * HEADS * NCHUNK, 256, 0, stream>>>(Qr, Kr, Vr, KV, Ksum, attn);

  // 5) output projection + bias via WMMA -> f32 out
  gemm_bf16<<<gemm_blocks, 256, 0, stream>>>(attn, Wot, bo, out);
}
